// MultiHeadAttention_68015102099715
// MI455X (gfx1250) — matrix operand order repaired and checked by reading, not compiled, not run on hardware
//
#include <hip/hip_runtime.h>
#include <hip/hip_bf16.h>

typedef __attribute__((ext_vector_type(16))) __bf16 v16bf;
typedef __attribute__((ext_vector_type(8)))  __bf16 v8bf;
typedef __attribute__((ext_vector_type(8)))  float  v8f;

#define B_  4
#define C_  1024
#define T_  1024
#define H_  16
#define HD_ 64

__device__ __forceinline__ v16bf cat8(v8bf lo, v8bf hi) {
  return __builtin_shufflevector(lo, hi, 0,1,2,3,4,5,6,7,8,9,10,11,12,13,14,15);
}

__device__ __forceinline__ v8f wmma_bf16(v16bf a, v16bf b, v8f c) {
  return __builtin_amdgcn_wmma_f32_16x16x32_bf16(false, a, false, b, (short)0, c,
                                                 false, false);
}

// Async DMA: 16 bytes global -> LDS, tracked by ASYNCcnt (CDNA5 path).
// LDS offset = low 32 bits of the flat shared-aperture address (ISA 10.2).
__device__ __forceinline__ void async_ld16(const __bf16* gptr, __bf16* lptr) {
  unsigned lds_off = (unsigned)(uintptr_t)lptr;
  asm volatile("global_load_async_to_lds_b128 %0, %1, off"
               :: "v"(lds_off), "v"(gptr) : "memory");
}
__device__ __forceinline__ void wait_async0() {
  asm volatile("s_wait_asynccnt 0x0" ::: "memory");
}

// ---------------------------------------------------------------- converts
__global__ void convert_f32_bf16(const float* __restrict__ src,
                                 __bf16* __restrict__ dst, int n) {
  int i = blockIdx.x * blockDim.x + threadIdx.x;
  if (i < n) dst[i] = (__bf16)src[i];
}

// x[b][c][t] f32 -> xT[b][t][c] bf16  (tiled 32x32 transpose)
__global__ __launch_bounds__(256) void transpose_convert(
    const float* __restrict__ X, __bf16* __restrict__ XT) {
  __shared__ float tile[32][33];
  const int b = blockIdx.z;
  const int t0 = blockIdx.x * 32, c0 = blockIdx.y * 32;
  const float* src = X + ((size_t)b * C_ + c0) * T_ + t0;
  for (int r = threadIdx.y; r < 32; r += 8)
    tile[r][threadIdx.x] = src[(size_t)r * T_ + threadIdx.x];
  __syncthreads();
  __bf16* dst = XT + ((size_t)b * T_ + t0) * C_ + c0;
  for (int r = threadIdx.y; r < 32; r += 8)
    dst[(size_t)r * C_ + threadIdx.x] = (__bf16)tile[threadIdx.x][r];
}

// ---------------------------------------------------------------- GEMM
// D[M,N] = A[M,K](bf16,row-major) x B (from BT[N,K] bf16 row-major) + bias[M]
// Block tile 64(M) x 128(N), 4 waves each 32x64.  K-step 32, double-buffered
// LDS filled by async direct-to-LDS DMA.
// OUT_MODE 0: bf16 Out[N][M] (transposed)   1: bf16 Out[M][N]   2: f32 Out[M][N]
template <int OUT_MODE>
__global__ __launch_bounds__(128) void gemm_bf16(
    const __bf16* __restrict__ A, const __bf16* __restrict__ BT,
    const float* __restrict__ bias, void* __restrict__ OutRaw,
    int M, int N, int K) {
  __shared__ __bf16 Als[2][64][40];
  __shared__ __bf16 Bls[2][128][40];
  const int z = blockIdx.z;
  BT += (size_t)z * N * K;
  const int m_blk = blockIdx.y * 64;
  const int n_blk = blockIdx.x * 128;
  const int tid  = threadIdx.x;
  const int lane = tid & 31;
  const int wave = tid >> 5;
  const int lr = lane & 15, hi = lane >> 4;
  const int wm = (wave >> 1) * 32;
  const int wn = (wave & 1) * 64;

  // staging assignment: A tile 64x32 (256 x 16B), B tile 128x32 (512 x 16B)
  const int arow = tid >> 1;            // 0..63
  const int aq0  = (tid & 1) * 2;       // quarters {0,1} or {2,3}
  const int brow = tid;                 // 0..127

  auto stage = [&](int buf, int kk) {
    const __bf16* ag = A + (size_t)(m_blk + arow) * K + kk;
#pragma unroll
    for (int i = 0; i < 2; ++i)
      async_ld16(ag + (aq0 + i) * 8, &Als[buf][arow][(aq0 + i) * 8]);
    const __bf16* bg = BT + (size_t)(n_blk + brow) * K + kk;
#pragma unroll
    for (int q = 0; q < 4; ++q)
      async_ld16(bg + q * 8, &Bls[buf][brow][q * 8]);
  };

  v8f acc[2][4] = {};
  stage(0, 0);
  wait_async0();
  __syncthreads();
  for (int kk = 0; kk < K; kk += 32) {
    const int buf = (kk >> 5) & 1;
    if (kk + 32 < K) stage(buf ^ 1, kk + 32);
    v16bf af[2], bfg[4];
#pragma unroll
    for (int i = 0; i < 2; ++i) {
      const __bf16* ap = &Als[buf][wm + i * 16 + lr][hi * 8];
      af[i] = cat8(*(const v8bf*)ap, *(const v8bf*)(ap + 16));
    }
#pragma unroll
    for (int j = 0; j < 4; ++j) {
      const __bf16* bp = &Bls[buf][wn + j * 16 + lr][hi * 8];
      bfg[j] = cat8(*(const v8bf*)bp, *(const v8bf*)(bp + 16));
    }
#pragma unroll
    for (int i = 0; i < 2; ++i)
#pragma unroll
      for (int j = 0; j < 4; ++j)
        acc[i][j] = wmma_bf16(af[i], bfg[j], acc[i][j]);
    wait_async0();
    __syncthreads();
  }

#pragma unroll
  for (int i = 0; i < 2; ++i) {
    const int m0 = m_blk + wm + i * 16 + hi * 8;
#pragma unroll
    for (int j = 0; j < 4; ++j) {
      const int n = n_blk + wn + j * 16 + lr;
      float vals[8];
#pragma unroll
      for (int e = 0; e < 8; ++e) vals[e] = acc[i][j][e] + bias[m0 + e];
      if (OUT_MODE == 0) {
        __bf16* O = (__bf16*)OutRaw + (size_t)z * N * M;
        v8bf o;
#pragma unroll
        for (int e = 0; e < 8; ++e) o[e] = (__bf16)vals[e];
        *(v8bf*)(O + (size_t)n * M + m0) = o;
      } else if (OUT_MODE == 1) {
        __bf16* O = (__bf16*)OutRaw + (size_t)z * M * N;
#pragma unroll
        for (int e = 0; e < 8; ++e) O[(size_t)(m0 + e) * N + n] = (__bf16)vals[e];
      } else {
        float* O = (float*)OutRaw + (size_t)z * M * N;
#pragma unroll
        for (int e = 0; e < 8; ++e) O[(size_t)(m0 + e) * N + n] = vals[e];
      }
    }
  }
}

// ---------------------------------------------------------------- RoPE (in place on qT,kT bf16 [B][T][C])
__global__ void rope_kernel(__bf16* __restrict__ qT, __bf16* __restrict__ kT) {
  const int idx = blockIdx.x * blockDim.x + threadIdx.x;  // b:2 t:10 h:4 j:4
  const int j = idx & 15;
  const int h = (idx >> 4) & 15;
  const int t = (idx >> 8) & 1023;
  const int b = idx >> 18;
  const float theta = __expf(-(float)j * (9.210340371976184f / 16.f)); // 10000^(-j/16)
  float sn, cs;
  __sincosf((float)t * theta, &sn, &cs);
  const size_t base = ((size_t)b * T_ + t) * C_ + h * HD_ + j;
  {
    float x1 = (float)qT[base], x2 = (float)qT[base + 16];
    qT[base]      = (__bf16)(x1 * cs - x2 * sn);
    qT[base + 16] = (__bf16)(x2 * cs + x1 * sn);
  }
  {
    float x1 = (float)kT[base], x2 = (float)kT[base + 16];
    kT[base]      = (__bf16)(x1 * cs - x2 * sn);
    kT[base + 16] = (__bf16)(x2 * cs + x1 * sn);
  }
}

// ---------------------------------------------------------------- attention
// qT,kT: [B][T][C] bf16 (roped).  vO: [B][C][T] bf16.  out attnT: [B][T][C] bf16
__global__ __launch_bounds__(128) void attn_fwd(
    const __bf16* __restrict__ qT, const __bf16* __restrict__ kT,
    const __bf16* __restrict__ vO, const int* __restrict__ mask,
    __bf16* __restrict__ attnT) {
  __shared__ __bf16 Ptile[4][16][40];
  const int b = blockIdx.z, h = blockIdx.y;
  const int wave = threadIdx.x >> 5;
  const int lane = threadIdx.x & 31;
  const int lr = lane & 15, hi = lane >> 4;
  const int trow0 = blockIdx.x * 64 + wave * 16;

  // Q fragments (A operand, rows t, K=d split into two 32-chunks)
  const __bf16* qp = qT + ((size_t)b * T_ + trow0 + lr) * C_ + h * HD_;
  v16bf qf[2];
#pragma unroll
  for (int f = 0; f < 2; ++f) {
    const __bf16* p = qp + f * 32 + hi * 8;
    qf[f] = cat8(*(const v8bf*)p, *(const v8bf*)(p + 16));
  }

  v8f mrow, lrow, oacc[4] = {};
#pragma unroll
  for (int e = 0; e < 8; ++e) { mrow[e] = -1e30f; lrow[e] = 0.f; }

  for (int s0 = 0; s0 < T_; s0 += 32) {
    v8f sc[2] = {};
#pragma unroll
    for (int st = 0; st < 2; ++st) {
      const int sb = s0 + st * 16;
      const __bf16* kp = kT + ((size_t)b * T_ + sb + lr) * C_ + h * HD_;
#pragma unroll
      for (int f = 0; f < 2; ++f) {
        const __bf16* p = kp + f * 32 + hi * 8;
        v16bf kf = cat8(*(const v8bf*)p, *(const v8bf*)(p + 16));
        sc[st] = wmma_bf16(qf[f], kf, sc[st]);
      }
    }
    // scale + mask + local row max
    v8f cmax;
#pragma unroll
    for (int e = 0; e < 8; ++e) cmax[e] = -1e30f;
#pragma unroll
    for (int st = 0; st < 2; ++st) {
      const int s = s0 + st * 16 + lr;
#pragma unroll
      for (int e = 0; e < 8; ++e) {
        float v = sc[st][e] * 0.125f;  // 1/sqrt(64)
        const int t = trow0 + hi * 8 + e;
        const int mv = mask[((size_t)b * T_ + t) * T_ + s];
        v = mv ? v : -10000.f;
        sc[st][e] = v;
        cmax[e] = fmaxf(cmax[e], v);
      }
    }
#pragma unroll
    for (int off = 1; off < 16; off <<= 1)
#pragma unroll
      for (int e = 0; e < 8; ++e)
        cmax[e] = fmaxf(cmax[e], __shfl_xor(cmax[e], off, 32));

    v8f mnew, factor;
#pragma unroll
    for (int e = 0; e < 8; ++e) {
      mnew[e] = fmaxf(mrow[e], cmax[e]);
      factor[e] = __expf(mrow[e] - mnew[e]);
    }
    v8f csum = {};
#pragma unroll
    for (int st = 0; st < 2; ++st) {
#pragma unroll
      for (int e = 0; e < 8; ++e) {
        float p = __expf(sc[st][e] - mnew[e]);
        csum[e] += p;
        Ptile[wave][hi * 8 + e][st * 16 + lr] = (__bf16)p;  // D-layout -> LDS
      }
    }
#pragma unroll
    for (int off = 1; off < 16; off <<= 1)
#pragma unroll
      for (int e = 0; e < 8; ++e) csum[e] += __shfl_xor(csum[e], off, 32);
#pragma unroll
    for (int e = 0; e < 8; ++e) {
      lrow[e] = lrow[e] * factor[e] + csum[e];
      mrow[e] = mnew[e];
    }
#pragma unroll
    for (int j = 0; j < 4; ++j)
#pragma unroll
      for (int e = 0; e < 8; ++e) oacc[j][e] *= factor[e];

    asm volatile("s_wait_dscnt 0x0" ::: "memory");  // P tile visible wave-wide

    const __bf16* pp = &Ptile[wave][lr][hi * 8];
    v16bf pf = cat8(*(const v8bf*)pp, *(const v8bf*)(pp + 16));  // A operand
#pragma unroll
    for (int j = 0; j < 4; ++j) {
      const __bf16* vp =
          vO + ((size_t)b * C_ + h * HD_ + j * 16 + lr) * T_ + s0 + hi * 8;
      v16bf vf = cat8(*(const v8bf*)vp, *(const v8bf*)(vp + 16));  // B operand
      oacc[j] = wmma_bf16(pf, vf, oacc[j]);
    }
    asm volatile("" ::: "memory");  // keep next iter's LDS stores after reads
  }

#pragma unroll
  for (int j = 0; j < 4; ++j)
#pragma unroll
    for (int e = 0; e < 8; ++e) {
      const float o = oacc[j][e] / lrow[e];
      const int t = trow0 + hi * 8 + e;
      attnT[((size_t)b * T_ + t) * C_ + h * HD_ + j * 16 + lr] = (__bf16)o;
    }
}

// ---------------------------------------------------------------- launcher
extern "C" void kernel_launch(void* const* d_in, const int* in_sizes, int n_in,
                              void* d_out, int out_size, void* d_ws,
                              size_t ws_size, hipStream_t stream) {
  (void)in_sizes; (void)n_in; (void)out_size; (void)ws_size;
  const float* x  = (const float*)d_in[0];
  const float* c  = (const float*)d_in[1];
  const int* mask = (const int*)d_in[2];
  const float* Wq = (const float*)d_in[3];
  const float* bq = (const float*)d_in[4];
  const float* Wk = (const float*)d_in[5];
  const float* bk = (const float*)d_in[6];
  const float* Wv = (const float*)d_in[7];
  const float* bv = (const float*)d_in[8];
  const float* Wo = (const float*)d_in[9];
  const float* bo = (const float*)d_in[10];

  char* ws = (char*)d_ws;
  size_t off = 0;
  auto alloc_bf = [&](size_t elems) {
    __bf16* p = (__bf16*)(ws + off);
    off += elems * sizeof(__bf16);
    return p;
  };
  __bf16* Wq_bf = alloc_bf((size_t)C_ * C_);
  __bf16* Wk_bf = alloc_bf((size_t)C_ * C_);
  __bf16* Wv_bf = alloc_bf((size_t)C_ * C_);
  __bf16* Wo_bf = alloc_bf((size_t)C_ * C_);
  __bf16* xT = alloc_bf((size_t)B_ * T_ * C_);
  __bf16* cT = alloc_bf((size_t)B_ * T_ * C_);
  __bf16* qT = alloc_bf((size_t)B_ * T_ * C_);
  __bf16* kT = alloc_bf((size_t)B_ * T_ * C_);
  __bf16* vO = alloc_bf((size_t)B_ * C_ * T_);
  __bf16* attnT = xT;  // x dead after Q projection; reuse

  const int nW = C_ * C_;
  convert_f32_bf16<<<(nW + 255) / 256, 256, 0, stream>>>(Wq, Wq_bf, nW);
  convert_f32_bf16<<<(nW + 255) / 256, 256, 0, stream>>>(Wk, Wk_bf, nW);
  convert_f32_bf16<<<(nW + 255) / 256, 256, 0, stream>>>(Wv, Wv_bf, nW);
  convert_f32_bf16<<<(nW + 255) / 256, 256, 0, stream>>>(Wo, Wo_bf, nW);

  dim3 tgrid(T_ / 32, C_ / 32, B_), tblk(32, 8);
  transpose_convert<<<tgrid, tblk, 0, stream>>>(x, xT);
  transpose_convert<<<tgrid, tblk, 0, stream>>>(c, cT);

  dim3 ggrid(T_ / 128, C_ / 64, B_);
  gemm_bf16<0><<<ggrid, 128, 0, stream>>>(Wq_bf, xT, bq, qT, C_, T_, C_);
  gemm_bf16<0><<<ggrid, 128, 0, stream>>>(Wk_bf, cT, bk, kT, C_, T_, C_);
  gemm_bf16<1><<<ggrid, 128, 0, stream>>>(Wv_bf, cT, bv, vO, C_, T_, C_);

  rope_kernel<<<(B_ * T_ * H_ * 16) / 256, 256, 0, stream>>>(qT, kT);

  dim3 agrid(T_ / 64, H_, B_);
  attn_fwd<<<agrid, 128, 0, stream>>>(qT, kT, vO, mask, attnT);

  gemm_bf16<2><<<ggrid, 128, 0, stream>>>(Wo_bf, attnT, bo, d_out, C_, T_, C_);
}
